// lstmVAE_feedback_68624987456256
// MI455X (gfx1250) — matrix operand order repaired and checked by reading, not compiled, not run on hardware
//
#include <hip/hip_runtime.h>
#include <cstdint>
#include <cstddef>

typedef __attribute__((ext_vector_type(8)))  __bf16 bf16x8;
typedef __attribute__((ext_vector_type(16))) __bf16 bf16x16;
typedef __attribute__((ext_vector_type(8)))  float  f32x8;

#define B_  256
#define L_  120
#define D_  192
#define Z_  512
#define H_  1024
#define FH_ 4096

// ---------------- fragment helpers (ISA 7.12.2 layouts, wave32) ----------------
// A-matrix 16x32 bf16: lane row M=l%16 holds K = hi*16 .. hi*16+15 (contiguous)
__device__ __forceinline__ bf16x16 load_a_frag(const __bf16* __restrict__ row, int hi) {
  return *(const bf16x16*)(row + hi * 16);
}

__device__ __forceinline__ float sigf(float x)  { return 1.0f / (1.0f + __expf(-x)); }
__device__ __forceinline__ float tanh_(float x) { return 1.0f - 2.0f / (__expf(2.0f * x) + 1.0f); }

__device__ __forceinline__ uint32_t pcg(uint32_t v) {
  v = v * 747796405u + 2891336453u;
  uint32_t w = ((v >> ((v >> 28) + 4u)) ^ v) * 277803737u;
  return (w >> 22) ^ w;
}

// ---------------- generic WMMA GEMM: out = act(A[M,K](bf16) @ W[N,K]^T + bias) ----------------
// grid = (M/16, N/64), block = 32 (one wave). Wave computes 16x64.
__global__ void __launch_bounds__(32)
wmma_gemm(const __bf16* __restrict__ A, long long As, int K,
          const __bf16* __restrict__ W, const float* __restrict__ bias,
          float* __restrict__ outf, long long ofs,
          __bf16* __restrict__ outbf, long long obs, int relu)
{
  const int lane = threadIdx.x;
  const int lo = lane & 15, hi = lane >> 4;
  const int m0 = blockIdx.x * 16;
  const int n0 = blockIdx.y * 64;
  f32x8 acc[4] = {};
  const __bf16* arow = A + (size_t)(m0 + lo) * As;
  for (int k0 = 0; k0 < K; k0 += 32) {
    bf16x16 af = load_a_frag(arow + k0, hi);
#pragma unroll
    for (int nt = 0; nt < 4; ++nt) {
      // B-matrix 32x16: lane holds N=lo, K-run = hi*16..+15 of W row (contiguous)
      const __bf16* wr = W + (size_t)(n0 + nt * 16 + lo) * K + k0 + hi * 16;
      bf16x16 bv = *(const bf16x16*)wr;
      acc[nt] = __builtin_amdgcn_wmma_f32_16x16x32_bf16(false, af, false, bv,
                                                        (short)0, acc[nt], false, false);
    }
  }
#pragma unroll
  for (int nt = 0; nt < 4; ++nt) {
    int col = n0 + nt * 16 + lo;
    float bvl = bias[col];
#pragma unroll
    for (int r = 0; r < 8; ++r) {
      int row = m0 + r + hi * 8;   // D layout: vgpr r, lanes>=16 -> M=r+8
      float v = acc[nt][r] + bvl;
      if (relu) v = fmaxf(v, 0.0f);
      if (outf)  outf [(size_t)row * ofs + col] = v;
      if (outbf) outbf[(size_t)row * obs + col] = (__bf16)v;
    }
  }
}

// ---------------- fused LSTM step ----------------
// gates[b, g*H+j] = a1[b,:]@Wih[g*H+j,:] + h[b,:]@Whh[g*H+j,:] + bih + bhh
// grid = (B/16, H/32), block = 32. Wave: 16 batch rows x 32 hidden units x 4 gates.
__global__ void __launch_bounds__(32)
lstm_step(const __bf16* __restrict__ a1, long long a1s, int K1,
          const __bf16* __restrict__ Wih,
          const __bf16* __restrict__ hinb,
          const __bf16* __restrict__ Whh,
          const float* __restrict__ bih, const float* __restrict__ bhh,
          const float* __restrict__ hin, const float* __restrict__ cin,
          float* __restrict__ hout, float* __restrict__ cout,
          __bf16* __restrict__ hbout,
          const int* __restrict__ lens, int t,
          float* __restrict__ pooled, int poff)
{
  const int lane = threadIdx.x;
  const int lo = lane & 15, hi = lane >> 4;
  const int b0 = blockIdx.x * 16;
  const int n0 = blockIdx.y * 32;
  f32x8 acc[4][2] = {};

  const __bf16* arow = a1 + (size_t)(b0 + lo) * a1s;
  for (int k0 = 0; k0 < K1; k0 += 32) {
    bf16x16 af = load_a_frag(arow + k0, hi);
#pragma unroll
    for (int g = 0; g < 4; ++g)
#pragma unroll
      for (int nt = 0; nt < 2; ++nt) {
        const __bf16* wr = Wih + (size_t)(g * H_ + n0 + nt * 16 + lo) * K1 + k0 + hi * 16;
        bf16x16 bv = *(const bf16x16*)wr;
        acc[g][nt] = __builtin_amdgcn_wmma_f32_16x16x32_bf16(false, af, false, bv,
                                                             (short)0, acc[g][nt], false, false);
      }
  }
  const __bf16* hrow = hinb + (size_t)(b0 + lo) * H_;
  for (int k0 = 0; k0 < H_; k0 += 32) {
    bf16x16 af = load_a_frag(hrow + k0, hi);
#pragma unroll
    for (int g = 0; g < 4; ++g)
#pragma unroll
      for (int nt = 0; nt < 2; ++nt) {
        const __bf16* wr = Whh + (size_t)(g * H_ + n0 + nt * 16 + lo) * H_ + k0 + hi * 16;
        bf16x16 bv = *(const bf16x16*)wr;
        acc[g][nt] = __builtin_amdgcn_wmma_f32_16x16x32_bf16(false, af, false, bv,
                                                             (short)0, acc[g][nt], false, false);
      }
  }

#pragma unroll
  for (int nt = 0; nt < 2; ++nt) {
    int hcol = n0 + nt * 16 + lo;
    float bi_i = bih[0 * H_ + hcol] + bhh[0 * H_ + hcol];
    float bi_f = bih[1 * H_ + hcol] + bhh[1 * H_ + hcol];
    float bi_g = bih[2 * H_ + hcol] + bhh[2 * H_ + hcol];
    float bi_o = bih[3 * H_ + hcol] + bhh[3 * H_ + hcol];
#pragma unroll
    for (int r = 0; r < 8; ++r) {
      int row = b0 + r + hi * 8;
      size_t idx = (size_t)row * H_ + hcol;
      float ig = sigf (acc[0][nt][r] + bi_i);
      float fg = sigf (acc[1][nt][r] + bi_f);
      float gg = tanh_(acc[2][nt][r] + bi_g);
      float og = sigf (acc[3][nt][r] + bi_o);
      float cold = cin[idx];
      float cn = fg * cold + ig * gg;
      float hn = og * tanh_(cn);
      bool act = (lens == nullptr) || (t < lens[row]);
      float hv = act ? hn : hin[idx];
      float cv = act ? cn : cold;
      hout[idx]  = hv;
      cout[idx]  = cv;
      hbout[idx] = (__bf16)hv;
      if (pooled && act) pooled[(size_t)row * (2 * H_) + poff + hcol] += hn;
    }
  }
}

// ---------------- small elementwise kernels ----------------
__global__ void cvt_bf16_k(const float* __restrict__ s, __bf16* __restrict__ d, int n) {
  int i = blockIdx.x * 256 + threadIdx.x; if (i < n) d[i] = (__bf16)s[i];
}
__global__ void zero_f32_k(float* p, int n) {
  int i = blockIdx.x * 256 + threadIdx.x; if (i < n) p[i] = 0.0f;
}
__global__ void copy_f32_k(const float* __restrict__ s, float* __restrict__ d, int n) {
  int i = blockIdx.x * 256 + threadIdx.x; if (i < n) d[i] = s[i];
}
__global__ void len_f32_k(const int* __restrict__ l, float* __restrict__ d, int n) {
  int i = blockIdx.x * 256 + threadIdx.x; if (i < n) d[i] = (float)l[i];
}
__global__ void ln_stats_k(const float* __restrict__ x, float* __restrict__ mean,
                           float* __restrict__ rstd) {
  __shared__ float s1[256], s2[256];
  int b = blockIdx.x, tid = threadIdx.x;
  const float* xb = x + (size_t)b * (L_ * D_);
  float a = 0.f, q = 0.f;
  for (int i = tid; i < L_ * D_; i += 256) { float v = xb[i]; a += v; q += v * v; }
  s1[tid] = a; s2[tid] = q; __syncthreads();
  for (int s = 128; s > 0; s >>= 1) {
    if (tid < s) { s1[tid] += s1[tid + s]; s2[tid] += s2[tid + s]; }
    __syncthreads();
  }
  if (tid == 0) {
    float m = s1[0] / (float)(L_ * D_);
    float v = s2[0] / (float)(L_ * D_) - m * m;
    mean[b] = m; rstd[b] = rsqrtf(v + 1e-5f);
  }
}
__global__ void ln_apply_k(const float* __restrict__ x, const float* __restrict__ mean,
                           const float* __restrict__ rstd, const float* __restrict__ w,
                           const float* __restrict__ bta, __bf16* __restrict__ o, int n) {
  int i = blockIdx.x * 256 + threadIdx.x; if (i >= n) return;
  int b = i / (L_ * D_); int r = i - b * (L_ * D_);
  o[i] = (__bf16)((x[i] - mean[b]) * rstd[b] * w[r] + bta[r]);
}
__global__ void pooled_fin_k(float* __restrict__ pooled, __bf16* __restrict__ pb,
                             const int* __restrict__ len, int n) {
  int i = blockIdx.x * 256 + threadIdx.x; if (i >= n) return;
  int b = i >> 11;   // / 2048
  float v = pooled[i] / (float)len[b];
  pooled[i] = v; pb[i] = (__bf16)v;
}
__global__ void z_k(const float* __restrict__ mu, const float* __restrict__ lv,
                    float* __restrict__ z, __bf16* __restrict__ zb, int n) {
  int i = blockIdx.x * 256 + threadIdx.x; if (i >= n) return;
  uint32_t r1 = pcg(2u * (uint32_t)i + 1u), r2 = pcg(2u * (uint32_t)i + 2u);
  float u1 = ((float)r1 + 1.0f) * (1.0f / 4294967296.0f);
  float u2 = (float)r2 * (1.0f / 4294967296.0f);
  float eps = sqrtf(-2.0f * __logf(u1)) * __cosf(6.28318530718f * u2);
  float zv = eps * __expf(0.5f * lv[i]) + mu[i];
  z[i] = zv; zb[i] = (__bf16)zv;
}

// ---------------- host driver ----------------
extern "C" void kernel_launch(void* const* d_in, const int* in_sizes, int n_in,
                              void* d_out, int out_size, void* d_ws, size_t ws_size,
                              hipStream_t stream) {
  const float* x     = (const float*)d_in[0];
  const int*   xlen  = (const int*)  d_in[1];
  const float* ln_w  = (const float*)d_in[2];
  const float* ln_b  = (const float*)d_in[3];
  const float* enc_w = (const float*)d_in[4];
  const float* enc_b = (const float*)d_in[5];
  const float* Wih_f = (const float*)d_in[6];
  const float* Whh_f = (const float*)d_in[7];
  const float* bih_f = (const float*)d_in[8];
  const float* bhh_f = (const float*)d_in[9];
  const float* Wih_b = (const float*)d_in[10];
  const float* Whh_b = (const float*)d_in[11];
  const float* bih_b = (const float*)d_in[12];
  const float* bhh_b = (const float*)d_in[13];
  const float* mu1_w = (const float*)d_in[14];
  const float* mu1_b = (const float*)d_in[15];
  const float* mu2_w = (const float*)d_in[16];
  const float* mu2_b = (const float*)d_in[17];
  const float* lv1_w = (const float*)d_in[18];
  const float* lv1_b = (const float*)d_in[19];
  const float* lv2_w = (const float*)d_in[20];
  const float* lv2_b = (const float*)d_in[21];
  const float* dz_w  = (const float*)d_in[22];
  const float* dz_b  = (const float*)d_in[23];
  const float* p1_w  = (const float*)d_in[24];
  const float* p1_b  = (const float*)d_in[25];
  const float* p2_w  = (const float*)d_in[26];
  const float* p2_b  = (const float*)d_in[27];
  const float* dWih  = (const float*)d_in[28];
  const float* dWhh  = (const float*)d_in[29];
  const float* dbih  = (const float*)d_in[30];
  const float* dbhh  = (const float*)d_in[31];
  const float* l1_w  = (const float*)d_in[32];
  const float* l1_b  = (const float*)d_in[33];
  const float* l2_w  = (const float*)d_in[34];
  const float* l2_b  = (const float*)d_in[35];

  float* out = (float*)d_out;
  const size_t x_n = (size_t)B_ * L_ * D_;
  float* out_x   = out;
  float* out_dec = out + x_n;
  float* out_len = out + 2 * x_n;
  float* out_mu  = out_len + B_;
  float* out_lv  = out_mu + (size_t)B_ * Z_;
  float* out_z   = out_lv + (size_t)B_ * Z_;

  char* wsp = (char*)d_ws;
  size_t off = 0;
  auto alloc = [&](size_t nbytes) -> void* {
    void* p = wsp + off; off = (off + nbytes + 255) & ~(size_t)255; return p;
  };
  auto abf  = [&](size_t n) { return (__bf16*)alloc(n * 2); };
  auto af32 = [&](size_t n) { return (float*) alloc(n * 4); };

  // bf16 weight copies (all stay L2-resident: ~52 MB total)
  __bf16* encw_b = abf((size_t)Z_ * D_);
  __bf16* wihf_b = abf((size_t)FH_ * Z_);
  __bf16* whhf_b = abf((size_t)FH_ * H_);
  __bf16* wihb_b = abf((size_t)FH_ * Z_);
  __bf16* whhb_b = abf((size_t)FH_ * H_);
  __bf16* mu1_bb = abf((size_t)H_ * 2 * H_);
  __bf16* mu2_bb = abf((size_t)Z_ * H_);
  __bf16* lv1_bb = abf((size_t)H_ * 2 * H_);
  __bf16* lv2_bb = abf((size_t)Z_ * H_);
  __bf16* dz_bb  = abf((size_t)Z_ * Z_);
  __bf16* p1_bb  = abf((size_t)Z_ * Z_);
  __bf16* p2_bb  = abf((size_t)D_ * Z_);
  __bf16* dwih_b = abf((size_t)FH_ * (Z_ + D_));
  __bf16* dwhh_b = abf((size_t)FH_ * H_);
  __bf16* l1_bb  = abf((size_t)Z_ * H_);
  __bf16* l2_bb  = abf((size_t)D_ * Z_);

  __bf16* xn_b = abf((size_t)B_ * L_ * D_);
  __bf16* xe_b = abf((size_t)B_ * L_ * Z_);   // row (b*L + t), stride Z
  float* mean = af32(B_);
  float* rstd = af32(B_);

  float *hF[2], *cF[2], *hB[2], *cB[2], *hD[2], *cD[2];
  __bf16 *hFb[2], *hBb[2], *hDb[2];
  for (int i = 0; i < 2; ++i) {
    hF[i] = af32((size_t)B_ * H_); cF[i] = af32((size_t)B_ * H_); hFb[i] = abf((size_t)B_ * H_);
    hB[i] = af32((size_t)B_ * H_); cB[i] = af32((size_t)B_ * H_); hBb[i] = abf((size_t)B_ * H_);
    hD[i] = af32((size_t)B_ * H_); cD[i] = af32((size_t)B_ * H_); hDb[i] = abf((size_t)B_ * H_);
  }
  float*  pooled  = af32((size_t)B_ * 2 * H_);
  __bf16* pooledb = abf((size_t)B_ * 2 * H_);
  __bf16* muh_b   = abf((size_t)B_ * H_);
  __bf16* lvh_b   = abf((size_t)B_ * H_);
  __bf16* z_b     = abf((size_t)B_ * Z_);
  __bf16* inp_b   = abf((size_t)B_ * (Z_ + D_));  // [xz | pose]
  __bf16* hidp_b  = abf((size_t)B_ * Z_);
  __bf16* hidd_b  = abf((size_t)B_ * Z_);

  auto ew = [&](int n) { return dim3((unsigned)((n + 255) / 256)); };
  auto cvt = [&](const float* s, __bf16* d, size_t n) {
    cvt_bf16_k<<<ew((int)n), 256, 0, stream>>>(s, d, (int)n);
  };
  auto gemm = [&](const __bf16* A, long long As, int K, const __bf16* W, const float* bias,
                  float* of, long long ofs, __bf16* ob, long long obs, int M, int N, int relu) {
    wmma_gemm<<<dim3((unsigned)(M / 16), (unsigned)(N / 64)), 32, 0, stream>>>(
        A, As, K, W, bias, of, ofs, ob, obs, relu);
  };
  auto lstm = [&](const __bf16* a1, long long a1s, int K1, const __bf16* Wih,
                  const __bf16* hinb, const __bf16* Whh, const float* bi, const float* bh,
                  const float* hin, const float* cin, float* hout, float* cout, __bf16* hbout,
                  const int* lens, int t, float* pl, int poff) {
    lstm_step<<<dim3(B_ / 16, H_ / 32), 32, 0, stream>>>(
        a1, a1s, K1, Wih, hinb, Whh, bi, bh, hin, cin, hout, cout, hbout, lens, t, pl, poff);
  };

  // 1) weight conversion
  cvt(enc_w, encw_b, (size_t)Z_ * D_);
  cvt(Wih_f, wihf_b, (size_t)FH_ * Z_);   cvt(Whh_f, whhf_b, (size_t)FH_ * H_);
  cvt(Wih_b, wihb_b, (size_t)FH_ * Z_);   cvt(Whh_b, whhb_b, (size_t)FH_ * H_);
  cvt(mu1_w, mu1_bb, (size_t)H_ * 2 * H_); cvt(mu2_w, mu2_bb, (size_t)Z_ * H_);
  cvt(lv1_w, lv1_bb, (size_t)H_ * 2 * H_); cvt(lv2_w, lv2_bb, (size_t)Z_ * H_);
  cvt(dz_w, dz_bb, (size_t)Z_ * Z_);      cvt(p1_w, p1_bb, (size_t)Z_ * Z_);
  cvt(p2_w, p2_bb, (size_t)D_ * Z_);
  cvt(dWih, dwih_b, (size_t)FH_ * (Z_ + D_)); cvt(dWhh, dwhh_b, (size_t)FH_ * H_);
  cvt(l1_w, l1_bb, (size_t)Z_ * H_);      cvt(l2_w, l2_bb, (size_t)D_ * Z_);

  // 2) passthrough outputs
  copy_f32_k<<<ew((int)x_n), 256, 0, stream>>>(x, out_x, (int)x_n);
  len_f32_k<<<1, 256, 0, stream>>>(xlen, out_len, B_);

  // 3) layernorm + encoder linear (WMMA): xe = relu(xn @ enc_w^T + enc_b), bf16
  ln_stats_k<<<B_, 256, 0, stream>>>(x, mean, rstd);
  ln_apply_k<<<ew((int)x_n), 256, 0, stream>>>(x, mean, rstd, ln_w, ln_b, xn_b, (int)x_n);
  gemm(xn_b, D_, D_, encw_b, enc_b, nullptr, 0, xe_b, Z_, B_ * L_, Z_, 1);

  // 4) zero initial state + pooled accumulator
  zero_f32_k<<<ew(B_ * 2 * H_), 256, 0, stream>>>(pooled, B_ * 2 * H_);
  float*  zf[] = { hF[0], cF[0], hB[0], cB[0], hD[0], cD[0] };
  __bf16* zb[] = { hFb[0], hBb[0], hDb[0] };
  for (float* p : zf) zero_f32_k<<<ew(B_ * H_), 256, 0, stream>>>(p, B_ * H_);
  for (__bf16* p : zb) zero_f32_k<<<ew(B_ * H_ / 2), 256, 0, stream>>>((float*)p, B_ * H_ / 2);

  // 5) bidirectional encoder LSTM (one step-kernel per t; h/c ping-pong)
  const long long xeStride = (long long)L_ * Z_;
  for (int t = 0; t < L_; ++t) {
    int in = t & 1, ot = in ^ 1;
    lstm(xe_b + (size_t)t * Z_, xeStride, Z_, wihf_b, hFb[in], whhf_b, bih_f, bhh_f,
         hF[in], cF[in], hF[ot], cF[ot], hFb[ot], xlen, t, pooled, 0);
  }
  for (int i = 0; i < L_; ++i) {
    int t = L_ - 1 - i;
    int in = i & 1, ot = in ^ 1;
    lstm(xe_b + (size_t)t * Z_, xeStride, Z_, wihb_b, hBb[in], whhb_b, bih_b, bhh_b,
         hB[in], cB[in], hB[ot], cB[ot], hBb[ot], xlen, t, pooled, H_);
  }

  // 6) pool, heads, reparameterize
  pooled_fin_k<<<ew(B_ * 2 * H_), 256, 0, stream>>>(pooled, pooledb, xlen, B_ * 2 * H_);
  gemm(pooledb, 2 * H_, 2 * H_, mu1_bb, mu1_b, nullptr, 0, muh_b, H_, B_, H_, 1);
  gemm(muh_b, H_, H_, mu2_bb, mu2_b, out_mu, Z_, nullptr, 0, B_, Z_, 0);
  gemm(pooledb, 2 * H_, 2 * H_, lv1_bb, lv1_b, nullptr, 0, lvh_b, H_, B_, H_, 1);
  gemm(lvh_b, H_, H_, lv2_bb, lv2_b, out_lv, Z_, nullptr, 0, B_, Z_, 0);
  z_k<<<ew(B_ * Z_), 256, 0, stream>>>(out_mu, out_lv, out_z, z_b, B_ * Z_);

  // 7) decoder init: xz = relu(z@dz^T) -> inp[:,0:512]; pose0 -> inp[:,512:704]
  gemm(z_b, Z_, Z_, dz_bb, dz_b, nullptr, 0, inp_b, Z_ + D_, B_, Z_, 1);
  gemm(inp_b, Z_ + D_, Z_, p1_bb, p1_b, nullptr, 0, hidp_b, Z_, B_, Z_, 1);
  gemm(hidp_b, Z_, Z_, p2_bb, p2_b, nullptr, 0, inp_b + Z_, Z_ + D_, B_, D_, 0);

  // 8) decoder loop with pose feedback
  for (int t = 0; t < L_; ++t) {
    int in = t & 1, ot = in ^ 1;
    lstm(inp_b, Z_ + D_, Z_ + D_, dwih_b, hDb[in], dwhh_b, dbih, dbhh,
         hD[in], cD[in], hD[ot], cD[ot], hDb[ot], nullptr, t, nullptr, 0);
    gemm(hDb[ot], H_, H_, l1_bb, l1_b, nullptr, 0, hidd_b, Z_, B_, Z_, 1);
    // pose -> dec[:, t, :] (f32, row stride L*D) and back into inp[:,512:] (bf16)
    gemm(hidd_b, Z_, Z_, l2_bb, l2_b, out_dec + (size_t)t * D_, (long long)L_ * D_,
         inp_b + Z_, Z_ + D_, B_, D_, 0);
  }
}
